// IndyLSTM_63677185131094
// MI455X (gfx1250) — matrix operand order repaired and checked by reading, not compiled, not run on hardware
//
#include <hip/hip_runtime.h>

// ---------------------------------------------------------------------------
// IndyLSTM for MI455X (gfx1250, wave32, WMMA + TDM)
//   Phase 1: fp32 -> f16 conversion of x and (transposed) gate weights
//   Phase 2: GEMM pre[B*T,4H] = x[B*T,I] * W[I,4H]
//            TDM (tensor_load_to_lds, padded rows) -> LDS double buffer ->
//            ds_load_b128 fragments -> v_wmma_f32_16x16x32_f16
//   Phase 3: elementwise-recurrent scan, parallel over B*H lanes
// ---------------------------------------------------------------------------

typedef __attribute__((ext_vector_type(16))) _Float16 v16h;
typedef __attribute__((ext_vector_type(8)))  _Float16 v8h;
typedef __attribute__((ext_vector_type(4)))  _Float16 v4h;
typedef __attribute__((ext_vector_type(8)))  float    v8f;
typedef __attribute__((ext_vector_type(4)))  float    v4f;
typedef __attribute__((ext_vector_type(4)))  unsigned int u32x4;
typedef __attribute__((ext_vector_type(8)))  int      i32x8;
typedef __attribute__((ext_vector_type(4)))  int      i32x4;

#if __has_builtin(__builtin_amdgcn_tensor_load_to_lds)
#define USE_TDM 1
#else
#define USE_TDM 0
#endif

namespace cfg {
constexpr int B = 32;
constexpr int T = 2048;
constexpr int I = 512;          // K
constexpr int H = 512;
constexpr int M = B * T;        // 65536 GEMM rows
constexpr int N = 4 * H;        // 2048  GEMM cols (4 gates)
constexpr int K = I;            // 512
}

// ---------------------------------------------------------------------------
// Phase 1a: x fp32 -> f16 (vectorized: float4 in, 4xf16 out)
// ---------------------------------------------------------------------------
__global__ void __launch_bounds__(256) cvt_x_f16(const float* __restrict__ x,
                                                 _Float16* __restrict__ xh,
                                                 int n4) {
    int i = blockIdx.x * blockDim.x + threadIdx.x;
    if (i >= n4) return;
    v4f v = *reinterpret_cast<const v4f*>(x + 4 * (size_t)i);
    v4h h;
    h.x = (_Float16)v.x; h.y = (_Float16)v.y;
    h.z = (_Float16)v.z; h.w = (_Float16)v.w;
    *reinterpret_cast<v4h*>(xh + 4 * (size_t)i) = h;
}

// ---------------------------------------------------------------------------
// Phase 1b: build WT[n][k] = W_g[k][h] in f16, n = g*H + h (K-major so the
// WMMA B-fragment per-lane loads are contiguous in k).
// ---------------------------------------------------------------------------
__global__ void __launch_bounds__(256) build_wt_f16(const float* __restrict__ Wf,
                                                    const float* __restrict__ Wi,
                                                    const float* __restrict__ Wo,
                                                    const float* __restrict__ Wc,
                                                    _Float16* __restrict__ WT) {
    using namespace cfg;
    int idx = blockIdx.x * blockDim.x + threadIdx.x;    // over (g,k,h)
    if (idx >= 4 * K * H) return;
    int h = idx % H;
    int k = (idx / H) % K;
    int g = idx / (K * H);
    const float* Wg = (g == 0) ? Wf : (g == 1) ? Wi : (g == 2) ? Wo : Wc;
    float w = Wg[(size_t)k * H + h];                    // coalesced read over h
    WT[((size_t)(g * H + h)) * K + k] = (_Float16)w;    // transposed write
}

// ---------------------------------------------------------------------------
// TDM: issue one 2-D f16 tile load (tile_k x tile_rows) global -> LDS with
// hardware row padding (16 DWORDs data + 4 DWORDs pad -> 80 B LDS row pitch,
// conflict-free for the ds_load_b128 fragment reads).
// D# packing per CDNA5 ISA 8.3/8.4.  6-arg builtin form (therock-10.0 HIP).
// ---------------------------------------------------------------------------
#if USE_TDM
__device__ __forceinline__ void tdm_load_tile_f16(unsigned int lds_off,
                                                  const _Float16* gptr,
                                                  unsigned int tile_k,
                                                  unsigned int tile_rows,
                                                  unsigned int row_stride_elems) {
    unsigned long long ga = (unsigned long long)(uintptr_t)gptr;
    u32x4 g0;
    g0.x = 1u;                                    // count=1, user descriptor
    g0.y = lds_off;                               // lds_addr [63:32]
    g0.z = (unsigned int)ga;                      // global_addr[31:0]
    g0.w = ((unsigned int)(ga >> 32) & 0x01FFFFFFu) | (2u << 30);  // [56:32] | type=2

    unsigned int d0 = row_stride_elems;           // tensor_dim0 (full row)
    unsigned int d1 = 1u << 20;                   // tensor_dim1: large (no clip)
    i32x8 g1;
    g1[0] = (int)((1u << 16)      // data_size = 2 bytes
                | (1u << 20)      // pad_enable
                | (3u << 22)      // pad_interval = 16 DWORDs (64 B)
                | (3u << 25));    // pad_amount   = 4 DWORDs (16 B) -> 80 B pitch
    g1[1] = (int)((d0 & 0xFFFFu) << 16);                          // dim0 lo
    g1[2] = (int)(((d0 >> 16) & 0xFFFFu) | ((d1 & 0xFFFFu) << 16));
    g1[3] = (int)(((d1 >> 16) & 0xFFFFu) | ((tile_k & 0xFFFFu) << 16));  // tile_dim0
    g1[4] = (int)(tile_rows & 0xFFFFu);           // tile_dim1 (tile_dim2 = 0)
    g1[5] = (int)row_stride_elems;                // tensor_dim0_stride[31:0]
    g1[6] = 0;                                    // stride0 hi, stride1 lo
    g1[7] = 0;                                    // stride1 hi
    i32x4 z4 = {0, 0, 0, 0};                      // groups 2/3: unused (2-D tile)
    i32x8 z8 = {0, 0, 0, 0, 0, 0, 0, 0};
    __builtin_amdgcn_tensor_load_to_lds(g0, g1, z4, z4, z8, 0);
}
#endif

// ---------------------------------------------------------------------------
// Phase 2: WMMA GEMM.  Wave tile 64x32 (4x2 fragments), 8 waves (2x4) per WG
// -> 128x128 workgroup tile.  Grid = (M/128)*(N/128) = 8192 blocks.
// K loop: 16 steps of 32, LDS double-buffered via TDM.
// ---------------------------------------------------------------------------
__global__ void __launch_bounds__(256, 2) gemm_wmma_f16(
        const _Float16* __restrict__ Xh,   // [M, K] row-major
        const _Float16* __restrict__ WT,   // [N, K] row-major (W transposed)
        float* __restrict__ Pre) {         // [M, N] row-major
    using namespace cfg;
    constexpr int WG_N_TILES = N / 128;    // 16

    const int tileM0 = (blockIdx.x / WG_N_TILES) * 128;
    const int tileN0 = (blockIdx.x % WG_N_TILES) * 128;
    const int wave   = threadIdx.x >> 5;   // 0..7
    const int lane   = threadIdx.x & 31;
    const int waveM  = wave >> 2;          // 0..1  (64 rows each)
    const int waveN  = wave & 3;           // 0..3  (32 cols each)

    const int lm = lane & 15;
    const int hi = lane >> 4;              // lane half-group

    // Per-lane fragment geometry (ISA 7.12.2, 16-bit wave32 layouts):
    //  A: row = lm, k-chunks {ka, ka+16}, ka = hi ? 8 : 0     (2 x 16B loads)
    //  B: col = lm, k-chunks {kb, kb+16}, kb = hi ? 8 : 0     (2 x 16B loads)
    const int ka = hi ? 8 : 0;
    const int kb = hi ? 8 : 0;

    v8f acc[4][2] = {};                    // 64x32 fp32 accumulators

#if USE_TDM
    constexpr int ROW_PITCH  = 80;               // 64 B data + 16 B TDM pad
    constexpr int TILE_BYTES = 128 * ROW_PITCH;  // 10 KB per tile
    __shared__ __align__(16) char smem[4 * TILE_BYTES];   // A0 B0 A1 B1
    const unsigned int lds0 = (unsigned int)(uintptr_t)&smem[0];

    auto issue = [&](int buf, int k0) {
        tdm_load_tile_f16(lds0 + (unsigned)(2 * buf + 0) * TILE_BYTES,
                          Xh + (size_t)tileM0 * K + k0, 32u, 128u, (unsigned)K);
        tdm_load_tile_f16(lds0 + (unsigned)(2 * buf + 1) * TILE_BYTES,
                          WT + (size_t)tileN0 * K + k0, 32u, 128u, (unsigned)K);
    };

    if (wave == 0) issue(0, 0);
    int buf = 0;
    for (int k0 = 0; k0 < K; k0 += 32) {
        if (wave == 0) __builtin_amdgcn_s_wait_tensorcnt(0);
        __syncthreads();                               // buffer ready + prev reads done
        if (wave == 0 && k0 + 32 < K) issue(buf ^ 1, k0 + 32);

        const char* As = smem + (2 * buf + 0) * TILE_BYTES;
        const char* Bs = smem + (2 * buf + 1) * TILE_BYTES;

        v16h afrag[4];
        v16h bfrag[2];
#pragma unroll
        for (int i = 0; i < 4; ++i) {
            const char* ar = As + (waveM * 64 + i * 16 + lm) * ROW_PITCH + ka * 2;
            reinterpret_cast<v8h*>(&afrag[i])[0] = *reinterpret_cast<const v8h*>(ar);
            reinterpret_cast<v8h*>(&afrag[i])[1] = *reinterpret_cast<const v8h*>(ar + 32);
        }
#pragma unroll
        for (int j = 0; j < 2; ++j) {
            const char* br = Bs + (waveN * 32 + j * 16 + lm) * ROW_PITCH + kb * 2;
            reinterpret_cast<v8h*>(&bfrag[j])[0] = *reinterpret_cast<const v8h*>(br);
            reinterpret_cast<v8h*>(&bfrag[j])[1] = *reinterpret_cast<const v8h*>(br + 32);
        }
#pragma unroll
        for (int i = 0; i < 4; ++i)
#pragma unroll
            for (int j = 0; j < 2; ++j)
                acc[i][j] = __builtin_amdgcn_wmma_f32_16x16x32_f16(
                    false, afrag[i], false, bfrag[j],
                    (short)0, acc[i][j], false, false);
        buf ^= 1;
    }
#else
    // Fallback: direct per-lane global fragment loads (round-1 path)
    const int wMg = tileM0 + waveM * 64;
    const int wNg = tileN0 + waveN * 32;
    for (int k0 = 0; k0 < K; k0 += 32) {
        v16h afrag[4];
        v16h bfrag[2];
#pragma unroll
        for (int i = 0; i < 4; ++i) {
            const _Float16* ab = Xh + (size_t)(wMg + i * 16 + lm) * K + k0 + ka;
            reinterpret_cast<v8h*>(&afrag[i])[0] = *reinterpret_cast<const v8h*>(ab);
            reinterpret_cast<v8h*>(&afrag[i])[1] = *reinterpret_cast<const v8h*>(ab + 16);
        }
#pragma unroll
        for (int j = 0; j < 2; ++j) {
            const _Float16* bb = WT + (size_t)(wNg + j * 16 + lm) * K + k0 + kb;
            reinterpret_cast<v8h*>(&bfrag[j])[0] = *reinterpret_cast<const v8h*>(bb);
            reinterpret_cast<v8h*>(&bfrag[j])[1] = *reinterpret_cast<const v8h*>(bb + 16);
        }
#pragma unroll
        for (int i = 0; i < 4; ++i)
#pragma unroll
            for (int j = 0; j < 2; ++j)
                acc[i][j] = __builtin_amdgcn_wmma_f32_16x16x32_f16(
                    false, afrag[i], false, bfrag[j],
                    (short)0, acc[i][j], false, false);
    }
#endif

    // C layout: VGPR r, lane l -> row (r + 8*(l>>4)), col (l&15).
    // Non-temporal: Pre is 512 MB streamed once, keep it out of L2's way.
#pragma unroll
    for (int i = 0; i < 4; ++i)
#pragma unroll
        for (int j = 0; j < 2; ++j) {
            const int col = tileN0 + waveN * 32 + j * 16 + lm;
#pragma unroll
            for (int r = 0; r < 8; ++r) {
                const int row = tileM0 + waveM * 64 + i * 16 + r + 8 * hi;
                __builtin_nontemporal_store(acc[i][j][r],
                                            &Pre[(size_t)row * N + col]);
            }
        }
}

// ---------------------------------------------------------------------------
// Phase 3: diagonal-recurrence scan.  One thread per (b,h); 2048 serial steps
// software-pipelined (t+1 loads issued before the t transcendental chain)
// plus global_prefetch_b8 a few steps ahead.
// ---------------------------------------------------------------------------
__device__ __forceinline__ float fast_sigmoid(float x) {
    return __frcp_rn(1.0f + __expf(-x));
}
__device__ __forceinline__ float fast_tanh(float x) {
    return 2.0f * __frcp_rn(1.0f + __expf(-2.0f * x)) - 1.0f;
}

__global__ void __launch_bounds__(256) indy_scan(
        const float* __restrict__ Pre,     // [B, T, 4, H]
        const float* __restrict__ u_f, const float* __restrict__ b_f,
        const float* __restrict__ u_i, const float* __restrict__ b_i,
        const float* __restrict__ u_o, const float* __restrict__ b_o,
        const float* __restrict__ u_c, const float* __restrict__ b_c,
        float* __restrict__ out) {         // [B,T,H] ++ h_t[B*H] ++ c_t[B*H]
    using namespace cfg;
    const int idx = blockIdx.x * blockDim.x + threadIdx.x;   // 0..B*H-1
    const int b = idx / H;
    const int h = idx % H;

    const float uf = u_f[h], bf = b_f[h];
    const float ui = u_i[h], bi = b_i[h];
    const float uo = u_o[h], bo = b_o[h];
    const float uc = u_c[h], bc = b_c[h];

    const float* __restrict__ prow = Pre + (size_t)b * T * 4 * H + h;
    float* __restrict__ orow = out + (size_t)b * T * H + h;

    float hp = 0.0f, c = 0.0f;

    float pf = prow[0 * H], pi = prow[1 * H], po = prow[2 * H], pz = prow[3 * H];

    for (int t = 0; t < T; ++t) {
        float nf = 0.f, ni = 0.f, no = 0.f, nz = 0.f;
        if (t + 1 < T) {
            const float* np = prow + (size_t)(t + 1) * 4 * H;
            nf = np[0 * H]; ni = np[1 * H]; no = np[2 * H]; nz = np[3 * H];
        }
        if (t + 8 < T)
            __builtin_prefetch(prow + (size_t)(t + 8) * 4 * H, 0, 0);

        const float gf = pf + hp * uf + bf;
        const float gi = pi + hp * ui + bi;
        const float go = po + hp * uo + bo;
        const float gz = pz + hp * uc + bc;

        const float f_t = fast_sigmoid(gf);
        const float i_t = fast_sigmoid(gi);
        const float o_t = fast_sigmoid(go);
        const float z_t = fast_tanh(gz);

        c  = f_t * c + i_t * z_t;
        hp = o_t * fast_tanh(c);

        orow[(size_t)t * H] = hp;
        pf = nf; pi = ni; po = no; pz = nz;
    }

    // reference returns INITIAL (zero) h_t, c_t -> zero the output tails
    out[(size_t)B * T * H + idx]         = 0.0f;
    out[(size_t)B * T * H + B * H + idx] = 0.0f;
}

// ---------------------------------------------------------------------------
// Host-side launcher
// ---------------------------------------------------------------------------
extern "C" void kernel_launch(void* const* d_in, const int* in_sizes, int n_in,
                              void* d_out, int out_size, void* d_ws, size_t ws_size,
                              hipStream_t stream) {
    using namespace cfg;
    (void)in_sizes; (void)n_in; (void)out_size; (void)ws_size;

    const float* x   = (const float*)d_in[0];
    const float* W_f = (const float*)d_in[1];
    const float* u_f = (const float*)d_in[2];
    const float* b_f = (const float*)d_in[3];
    const float* W_i = (const float*)d_in[4];
    const float* u_i = (const float*)d_in[5];
    const float* b_i = (const float*)d_in[6];
    const float* W_o = (const float*)d_in[7];
    const float* u_o = (const float*)d_in[8];
    const float* b_o = (const float*)d_in[9];
    const float* W_c = (const float*)d_in[10];
    const float* u_c = (const float*)d_in[11];
    const float* b_c = (const float*)d_in[12];
    float* out = (float*)d_out;

    // workspace layout
    char* ws = (char*)d_ws;
    float*    Pre = (float*)ws;                                   // M*N fp32 = 512 MB
    _Float16* Xh  = (_Float16*)(ws + (size_t)M * N * sizeof(float));        // 64 MB
    _Float16* WT  = (_Float16*)(ws + (size_t)M * N * sizeof(float)
                                   + (size_t)M * K * sizeof(_Float16));     // 2 MB

    // Phase 1: conversions
    {
        int n4 = (B * T * I) / 4;                                 // 8,388,608
        cvt_x_f16<<<(n4 + 255) / 256, 256, 0, stream>>>(x, Xh, n4);
        int nw = 4 * K * H;                                       // 1,048,576
        build_wt_f16<<<(nw + 255) / 256, 256, 0, stream>>>(W_f, W_i, W_o, W_c, WT);
    }

    // Phase 2: WMMA GEMM (128x128 workgroup tiles, TDM double buffering)
    {
        int grid = (M / 128) * (N / 128);                         // 8192
        gemm_wmma_f16<<<grid, 256, 0, stream>>>(Xh, WT, Pre);
    }

    // Phase 3: scan (one thread per (b,h))
    {
        int nthreads = B * H;                                     // 16384
        indy_scan<<<nthreads / 256, 256, 0, stream>>>(
            Pre, u_f, b_f, u_i, b_i, u_o, b_o, u_c, b_c, out);
    }
}
